// SelfAttention_17008070492194
// MI455X (gfx1250) — matrix operand order repaired and checked by reading, not compiled, not run on hardware
//
#include <hip/hip_runtime.h>

// ---------------------------------------------------------------------------
// Self-attention forward for MI455X (gfx1250, wave32, WMMA + TDM).
// B=8, T=1024, D=1024, H=16, DH=64.  Outputs: out [B,T,D] fp32 then
// attn_w [B,H,T,T] fp32, concatenated in d_out.
// GEMMs: v_wmma_f32_16x16x32_bf16 (bf16 in, fp32 accum), software-pipelined.
// Attention: K^T head staged into LDS by the Tensor Data Mover.
// ---------------------------------------------------------------------------

typedef __bf16 bf16;
typedef __attribute__((ext_vector_type(16))) bf16  v16bf;
typedef __attribute__((ext_vector_type(8)))  bf16  v8bf;
typedef __attribute__((ext_vector_type(8)))  float v8f;
typedef __attribute__((ext_vector_type(4)))  float v4f;
typedef __attribute__((ext_vector_type(4)))  unsigned int v4u;
typedef __attribute__((ext_vector_type(8)))  int   v8i;
typedef __attribute__((ext_vector_type(4)))  int   v4i;

#define B_  8
#define T_  1024
#define D_  1024
#define H_  16
#define DH_ 64
#define BT_ 8192
#define E3_ 3072
#define SCALE_ 0.125f  // DH^-0.5

// workspace offsets in bf16 elements
#define XB_OFF      ((size_t)0)          //  x as bf16          [BT, D]
#define WQKVT_OFF   ((size_t)8388608)    //  w_qkv^T bf16       [D, 3D]
#define WPROJT_OFF  ((size_t)11534336)   //  w_proj^T bf16      [D, D]
#define Q_OFF       ((size_t)12582912)   //  Q (pre-scaled)     [B,H,T,DH]
#define KT_OFF      ((size_t)20971520)   //  K transposed       [B,H,DH,T]
#define V_OFF       ((size_t)29360128)   //  V                  [B,H,T,DH]
#define AO_OFF      ((size_t)37748736)   //  attn out (concat)  [BT, D]

union AFrag { v16bf v; v8bf h[2]; bf16 e[16]; };

// A-frag: row-major A[M,K]; lane holds row M=lane&15; elems 0-7 at
// k0 + (lane>=16 ? 8 : 0), elems 8-15 at +16 (ISA 7.12.2, 16-bit A 16x32).
__device__ __forceinline__ v16bf load_a_frag(const bf16* __restrict__ A, int lda,
                                             int m0, int k0) {
  const int lane = threadIdx.x & 31;
  const int row  = m0 + (lane & 15);
  const int kb   = k0 + ((lane >> 4) << 3);
  AFrag f;
  f.h[0] = *(const v8bf*)(A + (size_t)row * lda + kb);
  f.h[1] = *(const v8bf*)(A + (size_t)row * lda + kb + 16);
  return f.v;
}

// B-frag: row-major Bm[K,N]; lane holds column N=n0+(lane&15); elems 0-7 at
// K=k0+(lane>=16 ? 8 : 0)+j, elems 8-15 at +16 (same k order as A-frag).
__device__ __forceinline__ v16bf load_b_frag(const bf16* __restrict__ Bm, int ldb,
                                             int k0, int n0) {
  const int lane = threadIdx.x & 31;
  const int n  = n0 + (lane & 15);
  const int kb = k0 + ((lane >> 4) << 3);
  AFrag f;
#pragma unroll
  for (int j = 0; j < 8; ++j) {
    f.e[j]     = Bm[(size_t)(kb + j) * ldb + n];
    f.e[8 + j] = Bm[(size_t)(kb + 16 + j) * ldb + n];
  }
  return f.v;
}

// One wave: 16x64 tile of A[M,K] * Bm[K,N]; explicitly software-pipelined so
// the next k-step's 5 fragment loads overlap the current 4 WMMAs.
__device__ __forceinline__ void gemm_16x64(const bf16* __restrict__ A, int lda,
                                           const bf16* __restrict__ Bm, int ldb,
                                           int m0, int n0, int K, v8f acc[4]) {
  const int lane = threadIdx.x & 31;
  v16bf a  = load_a_frag(A, lda, m0, 0);
  v16bf b0 = load_b_frag(Bm, ldb, 0, n0);
  v16bf b1 = load_b_frag(Bm, ldb, 0, n0 + 16);
  v16bf b2 = load_b_frag(Bm, ldb, 0, n0 + 32);
  v16bf b3 = load_b_frag(Bm, ldb, 0, n0 + 48);
  for (int k0 = 0; k0 < K; k0 += 32) {
    v16bf an = a, c0 = b0, c1 = b1, c2 = b2, c3 = b3;
    const int kn = k0 + 32;
    if (kn < K) {
      an = load_a_frag(A, lda, m0, kn);
      c0 = load_b_frag(Bm, ldb, kn, n0);
      c1 = load_b_frag(Bm, ldb, kn, n0 + 16);
      c2 = load_b_frag(Bm, ldb, kn, n0 + 32);
      c3 = load_b_frag(Bm, ldb, kn, n0 + 48);
      __builtin_prefetch(A + (size_t)(m0 + (lane & 15)) * lda + kn + 32, 0, 1);
      __builtin_prefetch(Bm + (size_t)(kn + 32 + lane) * ldb + n0, 0, 1);
    }
    acc[0] = __builtin_amdgcn_wmma_f32_16x16x32_bf16(false, a, false, b0, (short)0, acc[0], false, false);
    acc[1] = __builtin_amdgcn_wmma_f32_16x16x32_bf16(false, a, false, b1, (short)0, acc[1], false, false);
    acc[2] = __builtin_amdgcn_wmma_f32_16x16x32_bf16(false, a, false, b2, (short)0, acc[2], false, false);
    acc[3] = __builtin_amdgcn_wmma_f32_16x16x32_bf16(false, a, false, b3, (short)0, acc[3], false, false);
    a = an; b0 = c0; b1 = c1; b2 = c2; b3 = c3;
  }
}

// ---------------------------------------------------------------------------
// Tensor Data Mover: load one K^T head [64 rows x 1024 cols] bf16 (contiguous,
// row stride 1024 elems) into LDS at lds_off, padding +16B per 1KB of data
// (pad_enable=1, interval=256 dwords (code 7), amount=4 dwords (code 3)).
__device__ __forceinline__ void tdm_load_kt_head(const bf16* __restrict__ gptr,
                                                 unsigned lds_off) {
  const unsigned long long ga = (unsigned long long)(uintptr_t)gptr;
  v4u g0;
  g0[0] = 1u;                                   // count=1, user descriptor
  g0[1] = lds_off;                              // lds_addr (bytes)
  g0[2] = (unsigned)(ga & 0xFFFFFFFFu);         // global_addr[31:0]
  g0[3] = (unsigned)((ga >> 32) & 0x1FFFFFFu)   // global_addr[56:32]
        | (2u << 30);                           // type=2 ("image")
  v8i g1;
  g1[0] = (int)((1u << 16)      // data_size = 2 bytes
              | (1u << 20)      // pad_enable
              | (7u << 22)      // pad_interval = 256 dwords (1KB)
              | (3u << 25));    // pad_amount = 4 dwords (16B)
  g1[1] = (int)(1024u << 16);   // tensor_dim0 = 1024 (elems)
  g1[2] = (int)(64u << 16);     // tensor_dim1 = 64
  g1[3] = (int)(1024u << 16);   // tile_dim0 = 1024
  g1[4] = 64;                   // tile_dim1 = 64, tile_dim2 = 0
  g1[5] = 1024;                 // tensor_dim0_stride = 1024 (elems)
  g1[6] = 0;
  g1[7] = 0;
  v4i z4 = (v4i){0, 0, 0, 0};   // groups 2/3 unused (2D tensor)
#if __clang_major__ >= 23
  v8i z8 = (v8i){0, 0, 0, 0, 0, 0, 0, 0};
  __builtin_amdgcn_tensor_load_to_lds(g0, g1, z4, z4, z8, 0);
#else
  __builtin_amdgcn_tensor_load_to_lds(g0, g1, z4, z4, 0);
#endif
}

// LDS byte offset of padded K^T element (row, col): +16B per 1KB of raw data.
__device__ __forceinline__ unsigned kt_lds_off(int row, int col) {
  unsigned off = (unsigned)row * 2048u + (unsigned)col * 2u;
  return off + ((off >> 10) << 4);
}
#define KT_LDS_BYTES 133120u   // 64*2048 + 128 pads * 16B

// ---------------------------------------------------------------------------
// K0a: fp32 -> bf16 copy
__global__ void attn_cvt_kernel(const float* __restrict__ src,
                                bf16* __restrict__ dst, int n) {
  int i = blockIdx.x * blockDim.x + threadIdx.x;
  if (i < n) dst[i] = (bf16)src[i];
}

// K0b: fp32 [E, Din] -> bf16 [Din, E] (transpose + convert)
__global__ void attn_tcvt_kernel(const float* __restrict__ src,
                                 bf16* __restrict__ dst, int E, int Din) {
  int i = blockIdx.x * blockDim.x + threadIdx.x;
  if (i < E * Din) {
    int d = i / E;
    int e = i - d * E;
    dst[i] = (bf16)src[(size_t)e * Din + d];
  }
}

// ---------------------------------------------------------------------------
// K1: QKV GEMM.  C[bt, e] = Xb[bt, d] * WqkvT[d, e], e in [0, 3072).
__global__ void attn_qkv_kernel(const bf16* __restrict__ Xb,
                                const bf16* __restrict__ WqkvT,
                                bf16* __restrict__ Q,
                                bf16* __restrict__ KT,
                                bf16* __restrict__ V) {
  const int wave = threadIdx.x >> 5;
  const int lane = threadIdx.x & 31;
  const int m0 = blockIdx.x * 64 + wave * 16;
  const int n0 = blockIdx.y * 64;

  v8f acc[4];
#pragma unroll
  for (int t = 0; t < 4; ++t) acc[t] = (v8f){0,0,0,0,0,0,0,0};

  gemm_16x64(Xb, D_, WqkvT, E3_, m0, n0, D_, acc);

  const int colb = (lane < 16) ? lane : (lane - 16);
  const int mrb  = (lane < 16) ? 0 : 8;
#pragma unroll
  for (int t = 0; t < 4; ++t) {
#pragma unroll
    for (int i = 0; i < 8; ++i) {
      float v = acc[t][i];
      int m  = m0 + mrb + i;
      int e  = n0 + t * 16 + colb;
      int b  = m >> 10;
      int tk = m & 1023;
      if (e < 1024) {              // Q, pre-scaled
        int h = e >> 6, d = e & 63;
        Q[(((size_t)(b * H_ + h)) * T_ + tk) * DH_ + d] = (bf16)(v * SCALE_);
      } else if (e < 2048) {       // K stored transposed [DH, T]
        int f = e - 1024;
        int h = f >> 6, d = f & 63;
        KT[(((size_t)(b * H_ + h)) * DH_ + d) * T_ + tk] = (bf16)v;
      } else {                     // V [T, DH]
        int f = e - 2048;
        int h = f >> 6, d = f & 63;
        V[(((size_t)(b * H_ + h)) * T_ + tk) * DH_ + d] = (bf16)v;
      }
    }
  }
}

// ---------------------------------------------------------------------------
// K2: fused attention per (b,h).  TDM stages K^T head into LDS once per block;
// each wave owns 16 query rows: scores->LDS, row stats, softmax->attn_w, P*V.
#define SROW_ 1032  // 1024 + 8 pad floats per LDS score row

__global__ void attn_core_kernel(const bf16* __restrict__ Q,
                                 const bf16* __restrict__ KT,
                                 const bf16* __restrict__ V,
                                 float* __restrict__ attn_w,
                                 bf16* __restrict__ AO) {
  extern __shared__ char smem_raw[];   // [KT padded 133120B][scores 2*16*1032*4B]
  const int wave = threadIdx.x >> 5;
  const int lane = threadIdx.x & 31;
  const int bh = blockIdx.y;                 // b*H + h
  const int b  = bh >> 4;
  const int h  = bh & 15;
  const int q0 = blockIdx.x * 32 + wave * 16;

  char* ldsKT = smem_raw;
  float* sc = (float*)(smem_raw + KT_LDS_BYTES) + (size_t)wave * 16 * SROW_;

  const bf16* Qh  = Q  + (size_t)bh * T_ * DH_;  // [T, DH]
  const bf16* KTh = KT + (size_t)bh * DH_ * T_;  // [DH, T]
  const bf16* Vh  = V  + (size_t)bh * T_ * DH_;  // [T, DH]

  // ---- TDM: pull the whole K^T head (128KB) into LDS, shared by both waves
  if (wave == 0) {
    tdm_load_kt_head(KTh, (unsigned)(uintptr_t)(void*)ldsKT);
    __builtin_amdgcn_s_wait_tensorcnt(0);
  }
  __syncthreads();

  // Q tile fixed for this wave: 16 x 64 -> two A frags (k0 = 0, 32)
  v16bf qa0 = load_a_frag(Qh, DH_, q0, 0);
  v16bf qa1 = load_a_frag(Qh, DH_, q0, 32);

  const int colb = (lane < 16) ? lane : (lane - 16);
  const int mrb  = (lane < 16) ? 0 : 8;

  // ---- pass 1: scores into LDS (B-frags read from LDS-resident K^T) ----
  for (int n0 = 0; n0 < T_; n0 += 16) {
    v8f s = (v8f){0,0,0,0,0,0,0,0};
    AFrag kf0, kf1;
#pragma unroll
    for (int j = 0; j < 16; ++j) {
      const int kk = ((lane >> 4) << 3) + (j & 7) + ((j >> 3) << 4);
      kf0.e[j] = *(const bf16*)(ldsKT + kt_lds_off(kk,      n0 + (lane & 15)));
      kf1.e[j] = *(const bf16*)(ldsKT + kt_lds_off(32 + kk, n0 + (lane & 15)));
    }
    v16bf kb0 = kf0.v;
    v16bf kb1 = kf1.v;
    s = __builtin_amdgcn_wmma_f32_16x16x32_bf16(false, qa0, false, kb0, (short)0, s, false, false);
    s = __builtin_amdgcn_wmma_f32_16x16x32_bf16(false, qa1, false, kb1, (short)0, s, false, false);
#pragma unroll
    for (int i = 0; i < 8; ++i)
      sc[(size_t)(mrb + i) * SROW_ + n0 + colb] = s[i];
  }
  __syncthreads();

  // ---- row stats: lane pair (r, r+16) splits the 1024 columns ----
  const int r    = lane & 15;
  const int half = lane >> 4;
  const float* rowp = sc + (size_t)r * SROW_ + half * 512;
  float mx = -3.0e38f;
  for (int j = 0; j < 512; j += 4) {
    v4f x = *(const v4f*)(rowp + j);
    mx = fmaxf(mx, fmaxf(fmaxf(x[0], x[1]), fmaxf(x[2], x[3])));
  }
  mx = fmaxf(mx, __shfl_xor(mx, 16));
  float sum = 0.0f;
  for (int j = 0; j < 512; j += 4) {
    v4f x = *(const v4f*)(rowp + j);
    sum += __expf(x[0] - mx) + __expf(x[1] - mx) + __expf(x[2] - mx) + __expf(x[3] - mx);
  }
  sum += __shfl_xor(sum, 16);
  const float inv = 1.0f / sum;

  // ---- pass 2: P to attn_w (written exactly once) + O = P*V ----
  float* attnRow = attn_w + ((size_t)bh * T_ + (q0 + r)) * T_;
  v8f acc[4];
#pragma unroll
  for (int t = 0; t < 4; ++t) acc[t] = (v8f){0,0,0,0,0,0,0,0};

  for (int n0 = 0; n0 < T_; n0 += 32) {
    const int kb = n0 + (half << 3);
    AFrag a;
    float pb[16];
#pragma unroll
    for (int j = 0; j < 8; ++j) {           // K = kb + j
      float p = __expf(sc[(size_t)r * SROW_ + kb + j] - mx) * inv;
      a.e[j] = (bf16)p;
      pb[j] = p;
    }
#pragma unroll
    for (int j = 0; j < 8; ++j) {           // K = kb + 16 + j
      float p = __expf(sc[(size_t)r * SROW_ + kb + 16 + j] - mx) * inv;
      a.e[8 + j] = (bf16)p;
      pb[8 + j] = p;
    }
    *(v4f*)(attnRow + kb)      = (v4f){pb[0],  pb[1],  pb[2],  pb[3]};
    *(v4f*)(attnRow + kb + 4)  = (v4f){pb[4],  pb[5],  pb[6],  pb[7]};
    *(v4f*)(attnRow + kb + 16) = (v4f){pb[8],  pb[9],  pb[10], pb[11]};
    *(v4f*)(attnRow + kb + 20) = (v4f){pb[12], pb[13], pb[14], pb[15]};
#pragma unroll
    for (int t = 0; t < 4; ++t) {
      v16bf vb = load_b_frag(Vh, DH_, n0, t * 16);
      acc[t] = __builtin_amdgcn_wmma_f32_16x16x32_bf16(
          false, a.v, false, vb, (short)0, acc[t], false, false);
    }
  }

  // epilogue: concat heads -> AO [BT, D] bf16
#pragma unroll
  for (int t = 0; t < 4; ++t) {
#pragma unroll
    for (int i = 0; i < 8; ++i) {
      int m    = q0 + mrb + i;
      int dcol = h * DH_ + t * 16 + colb;
      AO[((size_t)b * T_ + m) * D_ + dcol] = (bf16)acc[t][i];
    }
  }
}

// ---------------------------------------------------------------------------
// K3: output projection.  out[bt, n] = AO[bt, d] * WprojT[d, n], fp32 store.
__global__ void attn_proj_kernel(const bf16* __restrict__ AO,
                                 const bf16* __restrict__ WprojT,
                                 float* __restrict__ out) {
  const int wave = threadIdx.x >> 5;
  const int lane = threadIdx.x & 31;
  const int m0 = blockIdx.x * 64 + wave * 16;
  const int n0 = blockIdx.y * 64;

  v8f acc[4];
#pragma unroll
  for (int t = 0; t < 4; ++t) acc[t] = (v8f){0,0,0,0,0,0,0,0};

  gemm_16x64(AO, D_, WprojT, D_, m0, n0, D_, acc);

  const int colb = (lane < 16) ? lane : (lane - 16);
  const int mrb  = (lane < 16) ? 0 : 8;
#pragma unroll
  for (int t = 0; t < 4; ++t) {
#pragma unroll
    for (int i = 0; i < 8; ++i)
      out[(size_t)(m0 + mrb + i) * D_ + n0 + t * 16 + colb] = acc[t][i];
  }
}

// ---------------------------------------------------------------------------
extern "C" void kernel_launch(void* const* d_in, const int* in_sizes, int n_in,
                              void* d_out, int out_size, void* d_ws, size_t ws_size,
                              hipStream_t stream) {
  const float* x      = (const float*)d_in[0];
  const float* w_qkv  = (const float*)d_in[1];
  const float* w_proj = (const float*)d_in[2];

  float* out    = (float*)d_out;                          // [B,T,D]
  float* attn_w = out + (size_t)BT_ * D_;                 // [B,H,T,T]

  bf16* ws     = (bf16*)d_ws;
  bf16* Xb     = ws + XB_OFF;
  bf16* WqkvT  = ws + WQKVT_OFF;
  bf16* WprojT = ws + WPROJT_OFF;
  bf16* Q      = ws + Q_OFF;
  bf16* KT     = ws + KT_OFF;
  bf16* V      = ws + V_OFF;
  bf16* AO     = ws + AO_OFF;

  // K0: conversions / weight transposes
  attn_cvt_kernel<<<(BT_ * D_) / 256, 256, 0, stream>>>(x, Xb, BT_ * D_);
  attn_tcvt_kernel<<<(E3_ * D_) / 256, 256, 0, stream>>>(w_qkv, WqkvT, E3_, D_);
  attn_tcvt_kernel<<<(D_ * D_) / 256, 256, 0, stream>>>(w_proj, WprojT, D_, D_);

  // K1: QKV projection (M=8192, N=3072, K=1024)
  attn_qkv_kernel<<<dim3(BT_ / 64, E3_ / 64), 128, 0, stream>>>(Xb, WqkvT, Q, KT, V);

  // K2: fused attention; 2 waves/block; LDS = padded K^T head + score tiles
  size_t smem_bytes = (size_t)KT_LDS_BYTES + (size_t)2 * 16 * SROW_ * sizeof(float);
  attn_core_kernel<<<dim3(T_ / 32, B_ * H_), 64, smem_bytes, stream>>>(
      Q, KT, V, attn_w, AO);

  // K3: output projection (M=8192, N=1024, K=1024)
  attn_proj_kernel<<<dim3(BT_ / 64, D_ / 64), 128, 0, stream>>>(AO, WprojT, out);
}
